// CustomRNN_88871463289370
// MI455X (gfx1250) — matrix operand order repaired and checked by reading, not compiled, not run on hardware
//
#include <hip/hip_runtime.h>
#include <math.h>

#define HIDDEN   512
#define T_STEPS  256
#define HORIZON  24
#define M_ROWS   16          // batch rows per workgroup
#define HPAD     520         // padded LDS row stride (bf16 elems): 1040B = 16B-aligned, 4 banks apart per row

typedef __attribute__((ext_vector_type(16))) __bf16       v16bf;
typedef __attribute__((ext_vector_type(8)))  float        v8f;
typedef __attribute__((ext_vector_type(4)))  unsigned int v4u;

union FragU { v4u u2[2]; v16bf bf; };

__device__ __forceinline__ unsigned short f32_to_bf16(float f) {
  unsigned int u = __builtin_bit_cast(unsigned int, f);
  u += 0x7FFFu + ((u >> 16) & 1u);            // round-to-nearest-even
  return (unsigned short)(u >> 16);
}
__device__ __forceinline__ float bf16_to_f32(unsigned short s) {
  unsigned int u = ((unsigned int)s) << 16;
  return __builtin_bit_cast(float, u);
}
__device__ __forceinline__ float fast_tanh(float v) {
#if __has_builtin(__builtin_amdgcn_tanhf)
  return __builtin_amdgcn_tanhf(v);           // gfx1250 v_tanh_f32
#else
  return tanhf(v);
#endif
}

// -------------------------------------------------------------------------
// Pack Wh (f32 [H][H], row j = output neuron, col k = input) into bf16 WMMA
// B-fragment order. B is K x N with B[k][j] = Wh[j][k]. Fragment f = ct*16+kt
// (ct: 16-col tile of N, kt: 32-row tile of K). Within a fragment, lane n
// holds column j = ct*16 + (n&15), K range kt*32 + (n>>4)*16 + e, e=0..15,
// stored contiguously: dst[f*512 + n*16 + e]. One fragment = 1024 contiguous
// bytes -> two coalesced global_load_b128 per lane in the main loop.
// -------------------------------------------------------------------------
__global__ void pack_wh_kernel(const float* __restrict__ wh,
                               unsigned short* __restrict__ bp) {
  int idx  = blockIdx.x * blockDim.x + threadIdx.x;  // 0 .. 262143
  int e    = idx & 15;
  int lane = (idx >> 4) & 31;
  int f    = idx >> 9;                                // 0 .. 511
  int ct   = f >> 4;                                  // 0 .. 31
  int kt   = f & 15;                                  // 0 .. 15
  int j    = ct * 16 + (lane & 15);
  int k    = kt * 32 + (lane >> 4) * 16 + e;
  bp[idx]  = f32_to_bf16(wh[j * HIDDEN + k]);
}

// -------------------------------------------------------------------------
// Persistent-per-workgroup RNN: 16 batch rows per WG, 8 waves, each wave
// owns a 64-wide slab of the hidden dimension. 256 time steps in-kernel.
// -------------------------------------------------------------------------
__global__ __launch_bounds__(256)
void rnn_scan_kernel(const float* __restrict__ x,        // [B][T]
                     const float* __restrict__ wxw,      // [H]
                     const float* __restrict__ wxb,      // [H]
                     const float* __restrict__ whb,      // [H]
                     const float* __restrict__ fcw,      // [HORIZON][H]
                     const float* __restrict__ fcb,      // [HORIZON]
                     const unsigned short* __restrict__ bpack, // packed bf16 Wh
                     float* __restrict__ out) {          // [B][HORIZON]
  __shared__ __align__(16) unsigned short hbuf[M_ROWS * HPAD]; // bf16 hidden tile
  __shared__ float xbuf[M_ROWS];

  const int tid     = threadIdx.x;
  const int lane    = tid & 31;
  const int wave    = tid >> 5;            // 0..7
  const int half    = lane >> 4;           // 0/1
  const int lrow    = lane & 15;
  const int rowbase = blockIdx.x * M_ROWS; // global batch row base
  const int jw      = wave * 64;           // hidden-col slab base for this wave

  // h0 = 0
  for (int i = tid; i < M_ROWS * HPAD; i += blockDim.x) hbuf[i] = 0;
  if (tid < M_ROWS) xbuf[tid] = x[(rowbase + tid) * T_STEPS + 0];

  // Per-col-tile invariants: Wx weight and fused bias for j = jw + ct*16 + lrow
  float wxc[4], bic[4];
#pragma unroll
  for (int ct = 0; ct < 4; ++ct) {
    int j   = jw + ct * 16 + lrow;
    wxc[ct] = wxw[j];
    bic[ct] = wxb[j] + whb[j];
  }
  __syncthreads();

  for (int t = 0; t < T_STEPS; ++t) {
    v8f acc[4];
#pragma unroll
    for (int ct = 0; ct < 4; ++ct) {
      v8f z = {0.f, 0.f, 0.f, 0.f, 0.f, 0.f, 0.f, 0.f};
      acc[ct] = z;
    }

    // ---- GEMM: acc[ct] += h(16xK) * Wh^T(Kx16), K = 512 ----
#pragma unroll 2
    for (int kt = 0; kt < 16; ++kt) {
      // A fragment 16x32 bf16 from LDS: lane m<16 holds row m K{0..15},
      // lane m+16 holds row m K{16..31} (matches packed B k order).
      FragU a;
      const unsigned short* ap = &hbuf[lrow * HPAD + kt * 32 + half * 16];
      a.u2[0] = *(const v4u*)(ap);
      a.u2[1] = *(const v4u*)(ap + 8);
#pragma unroll
      for (int ct = 0; ct < 4; ++ct) {
        const int f = (jw >> 4) * 16 + ct * 16 + kt;   // ((wave*4+ct)*16 + kt)
        FragU b;
        const unsigned short* bpp = &bpack[f * 512 + lane * 16];
        b.u2[0] = *(const v4u*)(bpp);
        b.u2[1] = *(const v4u*)(bpp + 8);
        acc[ct] = __builtin_amdgcn_wmma_f32_16x16x32_bf16(
            false, a.bf, false, b.bf, (short)0, acc[ct], false, false);
      }
    }

    // Per-row x_t values (C layout: VGPR r -> row r + half*8)
    float xr[8];
#pragma unroll
    for (int r = 0; r < 8; ++r) xr[r] = xbuf[r + half * 8];

    __syncthreads();  // all LDS reads of old h / xbuf complete

    // ---- elementwise: tanh(acc + x*Wx + bias), write bf16 h back ----
#pragma unroll
    for (int ct = 0; ct < 4; ++ct) {
      const int j = jw + ct * 16 + lrow;
#pragma unroll
      for (int r = 0; r < 8; ++r) {
        float v = acc[ct][r] + xr[r] * wxc[ct] + bic[ct];
        v = fast_tanh(v);
        hbuf[(r + half * 8) * HPAD + j] = f32_to_bf16(v);
      }
    }
    if (tid < M_ROWS && (t + 1) < T_STEPS)
      xbuf[tid] = x[(rowbase + tid) * T_STEPS + (t + 1)];
    __syncthreads();  // new h / next x visible to everyone
  }

  // ---- head: out[b][o] = h_final . fc_w[o] + fc_b[o]  (24 outputs/row) ----
  const int m = tid & 15;
  for (int o = tid >> 4; o < HORIZON; o += 16) {
    float s = fcb[o];
    const float* w = &fcw[o * HIDDEN];
    const unsigned short* hr = &hbuf[m * HPAD];
    for (int k = 0; k < HIDDEN; k += 4) {
      s += bf16_to_f32(hr[k + 0]) * w[k + 0]
         + bf16_to_f32(hr[k + 1]) * w[k + 1]
         + bf16_to_f32(hr[k + 2]) * w[k + 2]
         + bf16_to_f32(hr[k + 3]) * w[k + 3];
    }
    out[(rowbase + m) * HORIZON + o] = s;
  }
}

extern "C" void kernel_launch(void* const* d_in, const int* in_sizes, int n_in,
                              void* d_out, int out_size, void* d_ws, size_t ws_size,
                              hipStream_t stream) {
  const float* x    = (const float*)d_in[0];  // [B][T]
  const float* wxw  = (const float*)d_in[1];  // [H]
  const float* wxb  = (const float*)d_in[2];  // [H]
  const float* whw  = (const float*)d_in[3];  // [H][H]
  const float* whb  = (const float*)d_in[4];  // [H]
  const float* fcw  = (const float*)d_in[5];  // [HORIZON][H]
  const float* fcb  = (const float*)d_in[6];  // [HORIZON]
  float* out        = (float*)d_out;

  unsigned short* bpack = (unsigned short*)d_ws;     // 512*512 bf16 = 512 KB

  const int B = in_sizes[0] / T_STEPS;               // 1024

  // 1) repack Wh into bf16 WMMA-fragment order (deterministic, every call)
  pack_wh_kernel<<<(HIDDEN * HIDDEN) / 256, 256, 0, stream>>>(whw, bpack);

  // 2) full in-kernel scan over 256 steps + head
  rnn_scan_kernel<<<B / M_ROWS, 256, 0, stream>>>(x, wxw, wxb, whb, fcw, fcb,
                                                  bpack, out);
}
